// EpisodicMemory_47004122087648
// MI455X (gfx1250) — compile-verified, matrix operand order checked by reading, not run on hardware
//
#include <hip/hip_runtime.h>
#include <math.h>
#include <stdint.h>

// ---------------------------------------------------------------------------
// EpisodicMemory for MI455X (gfx1250, wave32, WMMA).
//
//   Wf[r,c] = W[r,c] + W[r,c+2048]    (folds cat=[hL,hR,hL,hR] into [hL,hR])
//   per level: out = [hL|hR] @ Wf^T + b  -> gates -> hP  (fused epilogue)
//   A-matrix of the GEMM is the previous level buffer viewed as (M/2, 2048).
//   GEMM in bf16 / f32 accumulation via v_wmma_f32_16x16x32_bf16.
//
// Each wave: 32 rows x 16 cols of hP = 2 row sub-tiles x 3 gate columns
// = 6 WMMA accumulators; 5 fragment loads feed 6 WMMAs per K-step.
// Inner loop is software-pipelined (loads for k+32 issued before WMMAs on k).
// Out-of-tile A reads are unguarded (buffers padded; stores are guarded).
// ---------------------------------------------------------------------------

typedef __bf16 bf16_t;
typedef __attribute__((ext_vector_type(16))) __bf16 v16bf;
typedef __attribute__((ext_vector_type(8)))  __bf16 v8bf;
typedef __attribute__((ext_vector_type(8)))  float  v8f;

#define DDIM 1024      // D
#define KDIM 2048      // 2*D (fused A width)
#define NGATES 3072    // 3*D

// ---------------- prep kernels ----------------

__global__ __launch_bounds__(256) void fuse_weights_kernel(
    const float* __restrict__ W, bf16_t* __restrict__ Wf, int n /* 3072*2048 */) {
  int i = blockIdx.x * 256 + threadIdx.x;
  if (i < n) {
    int r = i / KDIM;
    int c = i - r * KDIM;
    const float* row = W + (size_t)r * (4 * DDIM);
    Wf[i] = (bf16_t)(row[c] + row[c + KDIM]);
  }
}

__global__ __launch_bounds__(256) void cvt_bf16_kernel(
    const float* __restrict__ src, bf16_t* __restrict__ dst, int n) {
  int i = blockIdx.x * 256 + threadIdx.x;
  if (i < n) dst[i] = (bf16_t)src[i];
}

// ---------------- WMMA fragment load ----------------
// 16-bit A-matrix 16x32 layout (ISA 7.12.2):
//   lanes 0-15 : M=lane,    K in {0..7}  and {16..23}
//   lanes 16-31: M=lane-16, K in {8..15} and {24..31}
// => per lane: two contiguous 16-byte chunks at +((laneHi)?8:0) and +16 more.
// B (32x16) addressed identically from Wf rows (B = Wf^T, contiguous in K).
__device__ __forceinline__ v16bf load_frag_b16(const bf16_t* p) {
  v8bf lo = *(const v8bf*)(p);
  v8bf hi = *(const v8bf*)(p + 16);
  v16bf r;
#pragma unroll
  for (int i = 0; i < 8; ++i) { r[i] = lo[i]; r[i + 8] = hi[i]; }
  return r;
}

__device__ __forceinline__ float sigmoidf_(float x) {
  return 1.0f / (1.0f + __expf(-x));
}

__device__ __forceinline__ v8f wmma_bf16(v16bf a, v16bf b, v8f c) {
  return __builtin_amdgcn_wmma_f32_16x16x32_bf16(false, a, false, b,
                                                 (short)0, c, false, false);
}

// ---------------- fused merge-level GEMM ----------------
// Block: 256 threads = 8 waves. Macro-tile: 64 rows x 64 hP-cols.
// wave w: rows [blockIdx.x*64 + (w>>2)*32, +32), cols [blockIdx.y*64+(w&3)*16, +16).
__global__ __launch_bounds__(256) void merge_level_kernel(
    const bf16_t* __restrict__ Ab,    // (rowsOut, 2048) bf16 = prev level
    const float*  __restrict__ curF,  // (2*rowsOut, 1024) f32 = prev level
    const bf16_t* __restrict__ Wf,    // (3072, 2048) bf16 fused weights
    const float*  __restrict__ bias,  // (3072,)
    float*  __restrict__ nextF,       // (rowsOut, 1024) f32
    bf16_t* __restrict__ nextB,       // (rowsOut, 1024) bf16
    float*  __restrict__ memBuf,      // (7, 1024) f32
    int rowsOut, int memoff) {
  const int lane   = threadIdx.x & 31;
  const int wave   = threadIdx.x >> 5;
  const int laneHi = lane >> 4;
  const int lane15 = lane & 15;

  const int rowBase = blockIdx.x * 64 + (wave >> 2) * 32;
  const int j0      = blockIdx.y * 64 + (wave & 3) * 16;
  const int kLane   = laneHi ? 8 : 0;

  const bf16_t* aPtr0 = Ab + (size_t)(rowBase + lane15)      * KDIM + kLane;
  const bf16_t* aPtr1 = Ab + (size_t)(rowBase + 16 + lane15) * KDIM + kLane;
  const bf16_t* bPtr0 = Wf + (size_t)(0 * DDIM + j0 + lane15) * KDIM + kLane;
  const bf16_t* bPtr1 = Wf + (size_t)(1 * DDIM + j0 + lane15) * KDIM + kLane;
  const bf16_t* bPtr2 = Wf + (size_t)(2 * DDIM + j0 + lane15) * KDIM + kLane;

  v8f aI0 = {}, aF0 = {}, aO0 = {};   // rows [rowBase,    +16)
  v8f aI1 = {}, aF1 = {}, aO1 = {};   // rows [rowBase+16, +16)

  // prologue: first fragment set
  v16bf a0 = load_frag_b16(aPtr0);
  v16bf a1 = load_frag_b16(aPtr1);
  v16bf b0 = load_frag_b16(bPtr0);
  v16bf b1 = load_frag_b16(bPtr1);
  v16bf b2 = load_frag_b16(bPtr2);

#pragma unroll 2
  for (int k = 0; k < KDIM; k += 32) {
    // issue next iteration's loads first (one harmless extra block at the
    // end: every workspace buffer is padded, so this never faults)
    const int kn = k + 32;
    v16bf a0n = load_frag_b16(aPtr0 + kn);
    v16bf a1n = load_frag_b16(aPtr1 + kn);
    v16bf b0n = load_frag_b16(bPtr0 + kn);
    v16bf b1n = load_frag_b16(bPtr1 + kn);
    v16bf b2n = load_frag_b16(bPtr2 + kn);

    aI0 = wmma_bf16(a0, b0, aI0);
    aF0 = wmma_bf16(a0, b1, aF0);
    aO0 = wmma_bf16(a0, b2, aO0);
    aI1 = wmma_bf16(a1, b0, aI1);
    aF1 = wmma_bf16(a1, b1, aF1);
    aO1 = wmma_bf16(a1, b2, aO1);

    a0 = a0n; a1 = a1n; b0 = b0n; b1 = b1n; b2 = b2n;
  }

  // ---- fused epilogue ----
  // C/D layout: element e -> row = rowTile + e + (laneHi?8:0), col = j0+lane15
  const int jcol = j0 + lane15;
  const float bI = bias[jcol];
  const float bF = bias[DDIM + jcol];
  const float bO = bias[2 * DDIM + jcol];

#pragma unroll
  for (int s = 0; s < 2; ++s) {
    const v8f& vI = s ? aI1 : aI0;
    const v8f& vF = s ? aF1 : aF0;
    const v8f& vO = s ? aO1 : aO0;
#pragma unroll
    for (int e = 0; e < 8; ++e) {
      int p = rowBase + s * 16 + e + laneHi * 8;
      if (p < rowsOut) {
        float ig = sigmoidf_(vI[e] + bI);
        float fg = sigmoidf_(vF[e] + bF);
        float oc = vO[e] + bO;
        float hL = curF[(size_t)(2 * p)     * DDIM + jcol];
        float hR = curF[(size_t)(2 * p + 1) * DDIM + jcol];
        float uP = fg * hL + (1.0f - fg) * hR + ig * tanhf(oc);
        float hP = tanhf(uP) * sigmoidf_(oc);
        size_t o = (size_t)p * DDIM + jcol;
        nextF[o] = hP;
        nextB[o] = (bf16_t)hP;
        if (memoff >= 0) memBuf[(size_t)(memoff + p) * DDIM + jcol] = hP;
      }
    }
  }
}

// ---------------- attention over MEM (64 x 7 -> softmax -> 64 x 1024) ------
__global__ __launch_bounds__(128) void attn_kernel(
    const float* __restrict__ ct,   // (64, 1024)
    const float* __restrict__ mem,  // (7, 1024)
    float* __restrict__ out) {      // (64, 1024)
  const int b = blockIdx.x;
  const int t = threadIdx.x;
  __shared__ float red[128];
  __shared__ float alpha[7];

  for (int k = 0; k < 7; ++k) {
    float s = 0.0f;
    for (int d = t; d < DDIM; d += 128)
      s += ct[(size_t)b * DDIM + d] * mem[(size_t)k * DDIM + d];
    red[t] = s;
    __syncthreads();
    for (int off = 64; off > 0; off >>= 1) {
      if (t < off) red[t] += red[t + off];
      __syncthreads();
    }
    if (t == 0) alpha[k] = red[0];
    __syncthreads();
  }
  if (t == 0) {
    float m = alpha[0];
    for (int k = 1; k < 7; ++k) m = fmaxf(m, alpha[k]);
    float s = 0.0f;
    for (int k = 0; k < 7; ++k) { alpha[k] = __expf(alpha[k] - m); s += alpha[k]; }
    float inv = 1.0f / s;
    for (int k = 0; k < 7; ++k) alpha[k] *= inv;
  }
  __syncthreads();
  for (int j = t; j < DDIM; j += 128) {
    float a = 0.0f;
#pragma unroll
    for (int k = 0; k < 7; ++k) a += alpha[k] * mem[(size_t)k * DDIM + j];
    out[(size_t)b * DDIM + j] = a;
  }
}

// ---------------- host launcher ----------------
extern "C" void kernel_launch(void* const* d_in, const int* in_sizes, int n_in,
                              void* d_out, int out_size, void* d_ws, size_t ws_size,
                              hipStream_t stream) {
  const float* queue = (const float*)d_in[0];  // 8192*1024
  const float* c_t   = (const float*)d_in[1];  // 64*1024
  const float* W     = (const float*)d_in[2];  // 3072*4096
  const float* b     = (const float*)d_in[3];  // 3072
  float* out = (float*)d_out;                  // 64*1024

  // workspace layout; each chunk padded by 4KB so pipelined over-reads and
  // partial-tile reads never fault. Total ~64 MB.
  const size_t PAD = 4096;
  uint8_t* ws = (uint8_t*)d_ws;
  size_t off = 0;
  bf16_t* Wf = (bf16_t*)(ws + off); off += (size_t)NGATES * KDIM * 2 + PAD;
  bf16_t* qb = (bf16_t*)(ws + off); off += (size_t)8192 * DDIM * 2 + PAD;
  float*  fA = (float*) (ws + off); off += (size_t)4096 * DDIM * 4 + PAD;
  float*  fB = (float*) (ws + off); off += (size_t)2048 * DDIM * 4 + PAD;
  bf16_t* bA = (bf16_t*)(ws + off); off += (size_t)4096 * DDIM * 2 + PAD;
  bf16_t* bB = (bf16_t*)(ws + off); off += (size_t)2048 * DDIM * 2 + PAD;
  float*  memBuf = (float*)(ws + off); off += (size_t)7 * DDIM * 4 + PAD;
  (void)ws_size; (void)in_sizes; (void)n_in; (void)out_size;

  {
    int n = NGATES * KDIM;
    fuse_weights_kernel<<<(n + 255) / 256, 256, 0, stream>>>(W, Wf, n);
  }
  {
    int n = 8192 * DDIM;
    cvt_bf16_kernel<<<(n + 255) / 256, 256, 0, stream>>>(queue, qb, n);
  }

  const float*  curF = queue;
  const bf16_t* curB = qb;
  for (int lvl = 0; lvl < 13; ++lvl) {
    int rows_in  = 8192 >> lvl;
    int rows_out = rows_in >> 1;
    int lvl_out  = lvl + 1;
    float*  nF = (lvl_out & 1) ? fA : fB;
    bf16_t* nB = (lvl_out & 1) ? bA : bB;
    // MEM = concat(layer13, layer12, layer11)
    int memoff = (lvl_out == 13) ? 0 : (lvl_out == 12) ? 1 : (lvl_out == 11) ? 3 : -1;
    dim3 grid((rows_out + 63) / 64, DDIM / 64);
    merge_level_kernel<<<grid, 256, 0, stream>>>(curB, curF, Wf, b,
                                                 nF, nB, memBuf, rows_out, memoff);
    curF = nF;
    curB = nB;
  }

  attn_kernel<<<64, 128, 0, stream>>>(c_t, memBuf, out);
}
